// Seq2Seq_30288109371795
// MI455X (gfx1250) — compile-verified, matrix operand order checked by reading, not run on hardware
//
#include <hip/hip_runtime.h>

#define Vv 32000
#define Ee 512
#define Hh 512
#define Bb 16
#define Kk 8
#define Ss 50
#define Tt 40
#define Nn 128
#define SOS 2
#define EOS 3
#define UNK 0
#define PAD 1
#define MINLEN 3
#define NEGC (-1e20f)

typedef __bf16 bf16;
typedef __attribute__((ext_vector_type(16))) __bf16 v16bf;
typedef __attribute__((ext_vector_type(8)))  float  v8f;

__device__ __forceinline__ float sigmoid_f(float x) { return 1.0f / (1.0f + __expf(-x)); }

// ---------------------------------------------------------------------------
// Generic elementwise helpers
// ---------------------------------------------------------------------------
__global__ void k_zero_f32(float* __restrict__ p, long n) {
  long i = (long)blockIdx.x * blockDim.x + threadIdx.x;
  long st = (long)gridDim.x * blockDim.x;
  for (; i < n; i += st) p[i] = 0.0f;
}

__global__ void k_f32_to_bf16(const float* __restrict__ in, bf16* __restrict__ out, long n) {
  long i = (long)blockIdx.x * blockDim.x + threadIdx.x;
  long st = (long)gridDim.x * blockDim.x;
  for (; i < n; i += st) out[i] = (bf16)in[i];
}

__global__ void k_gather_embed_bf16(const float* __restrict__ embed, const int* __restrict__ tok,
                                    bf16* __restrict__ out, int rows) {
  int i = blockIdx.x * blockDim.x + threadIdx.x;
  if (i >= rows * Ee) return;
  int r = i >> 9, c = i & 511;
  out[i] = (bf16)embed[(long)tok[r] * Ee + c];
}

// out[n,:] = in[n/K,:]
__global__ void k_expand_rows(const float* __restrict__ in, float* __restrict__ out) {
  int i = blockIdx.x * blockDim.x + threadIdx.x;
  if (i >= Nn * Hh) return;
  int n = i >> 9, c = i & 511;
  out[i] = in[(n / Kk) * Hh + c];
}

// out[n,:] = in[(n/K)*K,:]   (reference's hidden[flat_offsets])
__global__ void k_beam0_rows(const float* __restrict__ in, float* __restrict__ out) {
  int i = blockIdx.x * blockDim.x + threadIdx.x;
  if (i >= Nn * Hh) return;
  int n = i >> 9, c = i & 511;
  out[i] = in[(long)((n / Kk) * Kk) * Hh + c];
}

// ---------------------------------------------------------------------------
// WMMA bf16 GEMM: C[M x Nc] f32 = A[M x Kd] @ B[Nc x Kd]^T (+bias)(+tanh)(+bf16 copy)
// One wave per 16x16 output tile; 8 waves (consecutive n-tiles) per block.
// VGPR layouts per CDNA5 ISA 7.12.2 (wave32).
// ---------------------------------------------------------------------------
__global__ __launch_bounds__(256)
void k_gemm_bf16(const bf16* __restrict__ A, const bf16* __restrict__ Bm,
                 const float* __restrict__ bias, float* __restrict__ C,
                 bf16* __restrict__ Cbf, int M, int Nc, int Kd,
                 int lda, int ldb, int ldc, int act) {
  (void)M;
  int lane = threadIdx.x & 31;
  int wv   = threadIdx.x >> 5;
  int nt   = blockIdx.x * 8 + wv;       // wave-uniform
  int mt   = blockIdx.y;
  if (nt * 16 >= Nc) return;            // uniform per wave -> EXEC stays all-ones
  int l16 = lane & 15;
  int kh  = lane >> 4;                  // K-half select (ISA A/B bf16 layout)
  const bf16* Ap = A + (long)(mt * 16 + l16) * lda;
  const bf16* Bp = Bm + (long)(nt * 16 + l16) * ldb;
  v8f acc = {0.f, 0.f, 0.f, 0.f, 0.f, 0.f, 0.f, 0.f};
  for (int k0 = 0; k0 < Kd; k0 += 32) {
    union { v16bf v; uint4 q[2]; } ua, ub;
    ua.q[0] = *(const uint4*)(Ap + k0 + kh * 8);        // K = kh*8 .. kh*8+7
    ua.q[1] = *(const uint4*)(Ap + k0 + 16 + kh * 8);   // K = 16+kh*8 .. +7
    ub.q[0] = *(const uint4*)(Bp + k0 + kh * 8);
    ub.q[1] = *(const uint4*)(Bp + k0 + 16 + kh * 8);
    acc = __builtin_amdgcn_wmma_f32_16x16x32_bf16(false, ua.v, false, ub.v,
                                                  (short)0, acc, false, false);
  }
  int ncol = nt * 16 + l16;
  float bv = bias ? bias[ncol] : 0.0f;
#pragma unroll
  for (int r = 0; r < 8; ++r) {
    int m = mt * 16 + kh * 8 + r;       // C/D layout: lanes16-31 hold M=8..15
    float v = acc[r] + bv;
    if (act) v = tanhf(v);
    C[(long)m * ldc + ncol] = v;
    if (Cbf) Cbf[(long)m * ldc + ncol] = (bf16)v;
  }
}

// ---------------------------------------------------------------------------
// GRU gate update (encoder: masked by src_len, writes enc_outs row)
// ---------------------------------------------------------------------------
__global__ void k_gru_update(const float* __restrict__ gi, const float* __restrict__ gh,
                             const float* __restrict__ h_in, float* __restrict__ h_out,
                             bf16* __restrict__ h_out_bf, const int* __restrict__ src_len,
                             int t, float* __restrict__ enc_out_row, int rows) {
  int i = blockIdx.x * blockDim.x + threadIdx.x;
  if (i >= rows * Hh) return;
  int r = i >> 9, c = i & 511;
  long base = (long)r * 3 * Hh;
  float ir = gi[base + c], iz = gi[base + Hh + c], inn = gi[base + 2 * Hh + c];
  float hr = gh[base + c], hz = gh[base + Hh + c], hn  = gh[base + 2 * Hh + c];
  float rg = sigmoid_f(ir + hr);
  float z  = sigmoid_f(iz + hz);
  float nv = tanhf(inn + rg * hn);
  float hp = h_in[i];
  float hv = (1.0f - z) * nv + z * hp;
  if (src_len) hv = (t < src_len[r]) ? hv : hp;
  h_out[i] = hv;
  if (h_out_bf) h_out_bf[i] = (bf16)hv;
  if (enc_out_row) enc_out_row[i] = hv;
}

// ---------------------------------------------------------------------------
// Fused attention: scores + masked softmax + context; writes combin = bf16[h|ctx]
// One block (64 threads) per decoder row n.
// ---------------------------------------------------------------------------
__global__ void k_attention(const float* __restrict__ q, const float* __restrict__ enc_outs,
                            const int* __restrict__ src_len, const float* __restrict__ h_new,
                            bf16* __restrict__ combin) {
  int n = blockIdx.x;
  int b = n / Kk;
  int tid = threadIdx.x;
  __shared__ float sc[Ss];
  __shared__ float mx, sm;
  if (tid < Ss) {
    const float4* q4 = (const float4*)(q + (long)n * Hh);
    const float4* e4 = (const float4*)(enc_outs + ((long)tid * Bb + b) * Hh);
    float s = 0.0f;
    for (int h = 0; h < Hh / 4; ++h) {
      float4 a = q4[h], e = e4[h];
      s += a.x * e.x + a.y * e.y + a.z * e.z + a.w * e.w;
    }
    sc[tid] = (tid < src_len[b]) ? s : -1e9f;
  }
  __syncthreads();
  if (tid == 0) {
    float m = -3e38f;
    for (int s = 0; s < Ss; ++s) m = fmaxf(m, sc[s]);
    mx = m;
  }
  __syncthreads();
  if (tid < Ss) sc[tid] = __expf(sc[tid] - mx);
  __syncthreads();
  if (tid == 0) {
    float s = 0.0f;
    for (int j = 0; j < Ss; ++j) s += sc[j];
    sm = s;
  }
  __syncthreads();
  if (tid < Ss) sc[tid] /= sm;
  __syncthreads();
  for (int c = tid; c < Hh; c += blockDim.x) {
    float acc = 0.0f;
    for (int s = 0; s < Ss; ++s) acc += sc[s] * enc_outs[((long)s * Bb + b) * Hh + c];
    combin[(long)n * (2 * Hh) + c]      = (bf16)h_new[(long)n * Hh + c];
    combin[(long)n * (2 * Hh) + Hh + c] = (bf16)acc;
  }
}

// ---------------------------------------------------------------------------
// In-place log-softmax per row (V=32000), one 256-thread block per row
// ---------------------------------------------------------------------------
__global__ void k_logsoftmax(float* __restrict__ logits) {
  int n = blockIdx.x;
  float* row = logits + (long)n * Vv;
  __shared__ float red[256];
  int tid = threadIdx.x;
  float m = -3e38f;
  for (int j = tid; j < Vv; j += 256) m = fmaxf(m, row[j]);
  red[tid] = m;
  __syncthreads();
  for (int s = 128; s > 0; s >>= 1) { if (tid < s) red[tid] = fmaxf(red[tid], red[tid + s]); __syncthreads(); }
  m = red[0];
  __syncthreads();
  float sum = 0.0f;
  for (int j = tid; j < Vv; j += 256) sum += __expf(row[j] - m);
  red[tid] = sum;
  __syncthreads();
  for (int s = 128; s > 0; s >>= 1) { if (tid < s) red[tid] += red[tid + s]; __syncthreads(); }
  float lse = m + __logf(red[0]);
  __syncthreads();
  for (int j = tid; j < Vv; j += 256) row[j] -= lse;
}

// ---------------------------------------------------------------------------
// Per-row top-8 over V with UNK/EOS masking. One block per row.
// ---------------------------------------------------------------------------
__global__ void k_topk8(const float* __restrict__ logp, int t,
                        float* __restrict__ tk_sc, int* __restrict__ tk_idx) {
  int n = blockIdx.x;
  const float* row = logp + (long)n * Vv;
  float v[8]; int id[8];
#pragma unroll
  for (int j = 0; j < 8; ++j) { v[j] = -3e38f; id[j] = 0; }
  for (int j = threadIdx.x; j < Vv; j += 256) {
    float x = row[j];
    if (j == UNK) x = NEGC;
    if (j == EOS && t <= MINLEN) x = NEGC;
    if (x > v[7]) {
      int p = 7;
      while (p > 0 && x > v[p - 1]) { v[p] = v[p - 1]; id[p] = id[p - 1]; --p; }
      v[p] = x; id[p] = j;
    }
  }
  __shared__ float sv[256 * 8];
  __shared__ int   si[256 * 8];
#pragma unroll
  for (int j = 0; j < 8; ++j) { sv[threadIdx.x * 8 + j] = v[j]; si[threadIdx.x * 8 + j] = id[j]; }
  __syncthreads();
  if (threadIdx.x == 0) {
    int head[256];
    for (int q2 = 0; q2 < 256; ++q2) head[q2] = 0;
    for (int pick = 0; pick < 8; ++pick) {
      float bv = -3e38f; int bq = 0;
      for (int q2 = 0; q2 < 256; ++q2) {
        if (head[q2] < 8) {
          float c = sv[q2 * 8 + head[q2]];
          if (c > bv) { bv = c; bq = q2; }
        }
      }
      tk_sc[n * Kk + pick] = bv;
      tk_idx[n * Kk + pick] = si[bq * 8 + head[bq]];
      head[bq]++;
    }
  }
}

// ---------------------------------------------------------------------------
// Beam-search step (single block). Mirrors the reference semantics exactly.
// ---------------------------------------------------------------------------
__global__ void k_beam_step(int t, const float* __restrict__ tksc, const int* __restrict__ tkix,
                            float* __restrict__ bscore, int* __restrict__ tokens,
                            int* __restrict__ ending, float* __restrict__ dec_len,
                            int* __restrict__ last_tok, int* __restrict__ flat_idx) {
  __shared__ float s_cur[Nn];
  __shared__ int   s_tok[Nn];
  __shared__ int   s_fi[Nn];
  __shared__ float s_tlen[Nn];
  __shared__ int   s_endo[Nn];
  __shared__ float rowf[Nn];
  __shared__ int   rowi[Nn];
  int tid = threadIdx.x;
  if (tid < Nn) {
    int e = ending[tid];
    s_endo[tid] = e;
    s_tlen[tid] = e ? dec_len[tid] : dec_len[tid] + 1.0f;
  }
  __syncthreads();
  if (tid < Bb) {
    int b = tid;
    float tot[Kk * Kk], cur[Kk * Kk];
    int gtok[Kk * Kk];
    for (int k = 0; k < Kk; ++k) {
      int nsrc = b * Kk + k;
      float total = 0.0f;
      for (int tt2 = 0; tt2 < Tt; ++tt2) total += bscore[tt2 * Nn + nsrc];
      int endf = s_endo[nsrc];
      for (int j = 0; j < Kk; ++j) {
        float sc = tksc[nsrc * Kk + j];
        if (t > 1) sc += -(float)j * 0.1f;                 // diversity offset
        float fin = endf ? (-sc + ((j == 0) ? 0.0f : NEGC)) : 0.0f;
        tot[k * Kk + j]  = sc + total + fin;
        cur[k * Kk + j]  = sc + fin;
        gtok[k * Kk + j] = tkix[nsrc * Kk + j];
      }
    }
    for (int i = 0; i < Kk; ++i) {
      int best = 0; float bv = tot[0];
      for (int j = 1; j < Kk * Kk; ++j) { if (tot[j] > bv) { bv = tot[j]; best = j; } }
      tot[best] = -3e38f;
      s_fi[b * Kk + i]  = b * Kk + best / Kk;
      s_tok[b * Kk + i] = gtok[best];
      s_cur[b * Kk + i] = cur[best];
    }
  }
  __syncthreads();
  // permute token / score histories by flat_idx
  for (int tt2 = 0; tt2 < Tt; ++tt2) {
    if (tid < Nn) { rowf[tid] = bscore[tt2 * Nn + tid]; rowi[tid] = tokens[tt2 * Nn + tid]; }
    __syncthreads();
    if (tid < Nn) { bscore[tt2 * Nn + tid] = rowf[s_fi[tid]]; tokens[tt2 * Nn + tid] = rowi[s_fi[tid]]; }
    __syncthreads();
  }
  if (tid < Nn) {
    int fi = s_fi[tid];
    int endn = s_endo[fi];
    int lt = endn ? PAD : s_tok[tid];
    last_tok[tid] = lt;
    tokens[t * Nn + tid] = lt;
    bscore[t * Nn + tid] = (lt != PAD) ? s_cur[tid] : 0.0f;
    ending[tid] = (endn || (lt == EOS)) ? 1 : 0;
    dec_len[tid] = s_tlen[fi];
    flat_idx[tid] = fi;
  }
}

__global__ void k_init_state(float* __restrict__ bscore, int* __restrict__ tokens,
                             int* __restrict__ ending, float* __restrict__ dec_len,
                             int* __restrict__ last_tok) {
  int i = blockIdx.x * blockDim.x + threadIdx.x;
  if (i < Tt * Nn) {
    bscore[i] = (i < Nn) ? (((i % Kk) == 0) ? 0.0f : NEGC) : 0.0f;
    tokens[i] = (i < Nn) ? SOS : 0;
  }
  if (i < Nn) { ending[i] = 0; dec_len[i] = 0.0f; last_tok[i] = SOS; }
}

__global__ void k_finalize(const int* __restrict__ tokens, const float* __restrict__ bscore,
                           float* __restrict__ out) {
  int i = blockIdx.x * blockDim.x + threadIdx.x;
  if (i >= Tt * Nn) return;
  long base = (long)Tt * Nn * Vv;
  reinterpret_cast<int*>(out + base)[i] = tokens[i];
  out[base + Tt * Nn + i] = bscore[i];
}

// ---------------------------------------------------------------------------
// Host-side orchestration
// ---------------------------------------------------------------------------
extern "C" void kernel_launch(void* const* d_in, const int* in_sizes, int n_in,
                              void* d_out, int out_size, void* d_ws, size_t ws_size,
                              hipStream_t stream) {
  (void)in_sizes; (void)n_in; (void)out_size; (void)ws_size;
  const int*   src     = (const int*)d_in[0];
  const int*   srclen  = (const int*)d_in[1];
  const float* embed   = (const float*)d_in[2];
  const float* encWih  = (const float*)d_in[3];
  const float* encWhh  = (const float*)d_in[4];
  const float* encBih  = (const float*)d_in[5];
  const float* encBhh  = (const float*)d_in[6];
  const float* bridgeW = (const float*)d_in[7];
  const float* bridgeB = (const float*)d_in[8];
  const float* decWih  = (const float*)d_in[9];
  const float* decWhh  = (const float*)d_in[10];
  const float* decBih  = (const float*)d_in[11];
  const float* decBhh  = (const float*)d_in[12];
  const float* attnW   = (const float*)d_in[13];
  const float* combW   = (const float*)d_in[14];
  const float* combB   = (const float*)d_in[15];
  const float* outW    = (const float*)d_in[16];
  const float* outB    = (const float*)d_in[17];

  char* w = (char*)d_ws;
  size_t off = 0;
  auto alloc = [&](size_t bytes) -> void* {
    void* p = w + off;
    off = (off + bytes + 255) & ~(size_t)255;
    return p;
  };
  bf16*  bEncWih = (bf16*)alloc((size_t)3 * Hh * Ee * 2);
  bf16*  bEncWhh = (bf16*)alloc((size_t)3 * Hh * Hh * 2);
  bf16*  bDecWih = (bf16*)alloc((size_t)3 * Hh * Ee * 2);
  bf16*  bDecWhh = (bf16*)alloc((size_t)3 * Hh * Hh * 2);
  bf16*  bBridge = (bf16*)alloc((size_t)Hh * Hh * 2);
  bf16*  bAttn   = (bf16*)alloc((size_t)Hh * Hh * 2);
  bf16*  bComb   = (bf16*)alloc((size_t)Hh * 2 * Hh * 2);
  bf16*  bOutW   = (bf16*)alloc((size_t)Vv * Hh * 2);
  bf16*  xBf     = (bf16*)alloc((size_t)Nn * Ee * 2);
  float* hEnc    = (float*)alloc((size_t)Bb * Hh * 4);
  bf16*  hEncBf  = (bf16*)alloc((size_t)Bb * Hh * 2);
  float* gi      = (float*)alloc((size_t)Nn * 3 * Hh * 4);
  float* gh      = (float*)alloc((size_t)Nn * 3 * Hh * 4);
  float* encOuts = (float*)alloc((size_t)Ss * Bb * Hh * 4);
  float* bridged = (float*)alloc((size_t)Bb * Hh * 4);
  float* hA      = (float*)alloc((size_t)Nn * Hh * 4);
  float* hB      = (float*)alloc((size_t)Nn * Hh * 4);
  float* hNew    = (float*)alloc((size_t)Nn * Hh * 4);
  bf16*  hNewBf  = (bf16*)alloc((size_t)Nn * Hh * 2);
  bf16*  hCurBf  = (bf16*)alloc((size_t)Nn * Hh * 2);
  float* qBuf    = (float*)alloc((size_t)Nn * Hh * 4);
  bf16*  combin  = (bf16*)alloc((size_t)Nn * 2 * Hh * 2);
  float* oBuf    = (float*)alloc((size_t)Nn * Hh * 4);
  bf16*  oBf     = (bf16*)alloc((size_t)Nn * Hh * 2);
  float* logits  = (float*)alloc((size_t)Nn * Vv * 4);
  float* tkSc    = (float*)alloc((size_t)Nn * Kk * 4);
  int*   tkIdx   = (int*)alloc((size_t)Nn * Kk * 4);
  float* bScore  = (float*)alloc((size_t)Tt * Nn * 4);
  int*   tokens  = (int*)alloc((size_t)Tt * Nn * 4);
  int*   endingB = (int*)alloc((size_t)Nn * 4);
  float* decLen  = (float*)alloc((size_t)Nn * 4);
  int*   lastTok = (int*)alloc((size_t)Nn * 4);
  int*   flatIdx = (int*)alloc((size_t)Nn * 4);

  auto conv = [&](const float* in, bf16* outp, long n) {
    int g = (int)((n + 255) / 256); if (g > 4096) g = 4096;
    k_f32_to_bf16<<<g, 256, 0, stream>>>(in, outp, n);
  };
  auto gemm = [&](const bf16* A, const bf16* Bm, const float* bias, float* C, bf16* Cbf,
                  int M, int Nc, int Kd, int lda, int ldb, int ldc, int act) {
    dim3 g((Nc / 16 + 7) / 8, M / 16);
    k_gemm_bf16<<<g, 256, 0, stream>>>(A, Bm, bias, C, Cbf, M, Nc, Kd, lda, ldb, ldc, act);
  };

  // ---- weight precision conversion (fp32 -> bf16, once per call) ----
  conv(encWih, bEncWih, 3L * Hh * Ee);
  conv(encWhh, bEncWhh, 3L * Hh * Hh);
  conv(decWih, bDecWih, 3L * Hh * Ee);
  conv(decWhh, bDecWhh, 3L * Hh * Hh);
  conv(bridgeW, bBridge, (long)Hh * Hh);
  conv(attnW, bAttn, (long)Hh * Hh);
  conv(combW, bComb, (long)Hh * 2 * Hh);
  conv(outW, bOutW, (long)Vv * Hh);

  // ---- output_probs are zeros; also init state ----
  long outTotal = (long)Tt * Nn * Vv + 2L * Tt * Nn;
  k_zero_f32<<<8192, 256, 0, stream>>>((float*)d_out, outTotal);
  k_zero_f32<<<32, 256, 0, stream>>>(hEnc, (long)Bb * Hh);
  k_init_state<<<(Tt * Nn + 255) / 256, 256, 0, stream>>>(bScore, tokens, endingB, decLen, lastTok);

  // ---- encoder scan ----
  for (int t = 0; t < Ss; ++t) {
    k_gather_embed_bf16<<<(Bb * Ee + 255) / 256, 256, 0, stream>>>(embed, src + t * Bb, xBf, Bb);
    k_f32_to_bf16<<<(Bb * Hh + 255) / 256, 256, 0, stream>>>(hEnc, hEncBf, (long)Bb * Hh);
    gemm(xBf, bEncWih, encBih, gi, nullptr, Bb, 3 * Hh, Ee, Ee, Ee, 3 * Hh, 0);
    gemm(hEncBf, bEncWhh, encBhh, gh, nullptr, Bb, 3 * Hh, Hh, Hh, Hh, 3 * Hh, 0);
    k_gru_update<<<(Bb * Hh + 255) / 256, 256, 0, stream>>>(
        gi, gh, hEnc, hEnc, nullptr, srclen, t, encOuts + (long)t * Bb * Hh, Bb);
  }

  // ---- bridge + broadcast to beams ----
  k_f32_to_bf16<<<(Bb * Hh + 255) / 256, 256, 0, stream>>>(hEnc, hEncBf, (long)Bb * Hh);
  gemm(hEncBf, bBridge, bridgeB, bridged, nullptr, Bb, Hh, Hh, Hh, Hh, Hh, 1);
  k_expand_rows<<<(Nn * Hh + 255) / 256, 256, 0, stream>>>(bridged, hA);

  // ---- decoder + beam search ----
  for (int t = 1; t < Tt; ++t) {
    float* hcur  = (t & 1) ? hA : hB;
    float* hnext = (t & 1) ? hB : hA;
    k_gather_embed_bf16<<<(Nn * Ee + 255) / 256, 256, 0, stream>>>(embed, lastTok, xBf, Nn);
    k_f32_to_bf16<<<(Nn * Hh + 255) / 256, 256, 0, stream>>>(hcur, hCurBf, (long)Nn * Hh);
    gemm(xBf, bDecWih, decBih, gi, nullptr, Nn, 3 * Hh, Ee, Ee, Ee, 3 * Hh, 0);
    gemm(hCurBf, bDecWhh, decBhh, gh, nullptr, Nn, 3 * Hh, Hh, Hh, Hh, 3 * Hh, 0);
    k_gru_update<<<(Nn * Hh + 255) / 256, 256, 0, stream>>>(
        gi, gh, hcur, hNew, hNewBf, nullptr, 0, nullptr, Nn);
    gemm(hNewBf, bAttn, nullptr, qBuf, nullptr, Nn, Hh, Hh, Hh, Hh, Hh, 0);
    k_attention<<<Nn, 64, 0, stream>>>(qBuf, encOuts, srclen, hNew, combin);
    gemm(combin, bComb, combB, oBuf, oBf, Nn, Hh, 2 * Hh, 2 * Hh, 2 * Hh, Hh, 1);
    gemm(oBf, bOutW, outB, logits, nullptr, Nn, Vv, Hh, Hh, Hh, Vv, 0);
    k_logsoftmax<<<Nn, 256, 0, stream>>>(logits);
    k_topk8<<<Nn, 256, 0, stream>>>(logits, t, tkSc, tkIdx);
    k_beam_step<<<1, 256, 0, stream>>>(t, tkSc, tkIdx, bScore, tokens, endingB, decLen, lastTok, flatIdx);
    k_beam0_rows<<<(Nn * Hh + 255) / 256, 256, 0, stream>>>(hNew, hnext);
  }

  k_finalize<<<(Tt * Nn + 255) / 256, 256, 0, stream>>>(tokens, bScore, (float*)d_out);
}
